// GAT_Two_Layer_26371099197746
// MI455X (gfx1250) — matrix operand order repaired and checked by reading, not compiled, not run on hardware
//
#include <hip/hip_runtime.h>

// CDNA5 / gfx1250 GAT two-layer forward.
// WMMA f16->f32 for both node-feature GEMMs (one wave = 16-row strip x all n-tiles);
// atomic scatter for edge softmax/aggregation (the HBM-bound part).

typedef __attribute__((ext_vector_type(16))) _Float16    v16h;
typedef __attribute__((ext_vector_type(8)))  float       v8f;
typedef __attribute__((ext_vector_type(4)))  unsigned int u32x4;

#define F_IN     128
#define HID      32
#define HEADS    4
#define CLS      40
#define CLS_PAD  48
#define NEG_SLOPE 0.2f
#define TPB 256

// ---------------- precision conversion / weight packing ----------------

__global__ void k_cvt_f16(const float* __restrict__ src, _Float16* __restrict__ dst, int n) {
  int t = blockIdx.x * blockDim.x + threadIdx.x;
  if (t < n) dst[t] = (_Float16)src[t];
}

// W: [K][Nsrc] row-major  ->  WT: [Npad][K] (column-major of W), zero-padded columns
__global__ void k_packWT(const float* __restrict__ W, _Float16* __restrict__ WT,
                         int K, int Nsrc, int Npad) {
  int t = blockIdx.x * blockDim.x + threadIdx.x;
  if (t >= K * Npad) return;
  int n = t / K, k = t % K;
  WT[(size_t)n * K + k] = (n < Nsrc) ? (_Float16)W[(size_t)k * Nsrc + n] : (_Float16)0.0f;
}

// ---------------- WMMA GEMM:  C[M x (16*NT)] = A[M x 128] * B[128 x (16*NT)] ----------------
// A: row-major f16, lda = 128.  BT: B transposed (column-major), [16*NT][128] f16.
// One wave computes one 16-row strip across all NT n-tiles: the A fragment is
// loaded once per k-step and fed to NT consecutive WMMAs (A traffic / NT).

template<int NT>
__global__ void k_wmma_gemm(const _Float16* __restrict__ A, const _Float16* __restrict__ BT,
                            float* __restrict__ C, int ldc, int Nn, int mtiles) {
  int wid  = (blockIdx.x * blockDim.x + threadIdx.x) >> 5;   // wave id == m-tile
  int lane = threadIdx.x & 31;
  if (wid >= mtiles) return;
  int m0 = wid << 4;
  int rl = lane & 15;        // A: row in tile / B,C: column in tile
  int hf = lane >> 4;        // lane-half selector

  int arow = m0 + rl; if (arow >= Nn) arow = Nn - 1;         // clamp (loads only)

  v8f acc[NT];
#pragma unroll
  for (int nt = 0; nt < NT; ++nt) acc[nt] = {};

  const _Float16* arowp = A  + (size_t)arow * 128;
  const _Float16* bbase = BT + (size_t)rl * 128 + 8 * hf;    // + nt*16*128 + k0 per use

#pragma unroll
  for (int k0 = 0; k0 < 128; k0 += 32) {
    union { v16h v; u32x4 u[2]; } af;
    // A 16x32 f16 fragment: two contiguous 8-halfword chunks per lane
    const _Float16* ap = arowp + k0 + 8 * hf;
    af.u[0] = *(const u32x4*)(ap);
    af.u[1] = *(const u32x4*)(ap + 16);
#pragma unroll
    for (int nt = 0; nt < NT; ++nt) {
      union { v16h v; u32x4 u[2]; } bf;
      // B 32x16 f16 fragment: two contiguous 8-halfword chunks per lane of column (lane&15)
      const _Float16* bp = bbase + (size_t)nt * 16 * 128 + k0;
      bf.u[0] = *(const u32x4*)(bp);
      bf.u[1] = *(const u32x4*)(bp + 16);
      acc[nt] = __builtin_amdgcn_wmma_f32_16x16x32_f16(false, af.v, false, bf.v,
                                                       (short)0, acc[nt], false, false);
    }
  }

  // C/D layout: lane -> column nt*16+rl, VGPR j -> row j + 8*hf
  float* crow = C + (size_t)(m0 + 8 * hf) * ldc + rl;
  if (m0 + 16 <= Nn) {                       // interior strip: branchless stores
#pragma unroll
    for (int j = 0; j < 8; ++j)
#pragma unroll
      for (int nt = 0; nt < NT; ++nt)
        crow[(size_t)j * ldc + nt * 16] = acc[nt][j];
  } else {                                   // tail strip only
#pragma unroll
    for (int j = 0; j < 8; ++j)
      if (m0 + j + 8 * hf < Nn)
#pragma unroll
        for (int nt = 0; nt < NT; ++nt)
          crow[(size_t)j * ldc + nt * 16] = acc[nt][j];
  }
}

// ---------------- attention logits ----------------

__global__ void k_att1(const float* __restrict__ h, const float* __restrict__ a_src,
                       const float* __restrict__ a_dst, float* __restrict__ als,
                       float* __restrict__ ald, int Nn) {
  int t = blockIdx.x * blockDim.x + threadIdx.x;
  if (t >= Nn * HEADS) return;
  int n = t >> 2, hd = t & 3;
  const float* row = h + (size_t)n * F_IN + hd * HID;
  float s = 0.f, d = 0.f;
#pragma unroll
  for (int c = 0; c < HID; ++c) {
    float v = row[c];
    s += v * a_src[hd * HID + c];
    d += v * a_dst[hd * HID + c];
  }
  als[t] = s; ald[t] = d;
}

__global__ void k_att2(const float* __restrict__ h, const float* __restrict__ a_src,
                       const float* __restrict__ a_dst, float* __restrict__ als,
                       float* __restrict__ ald, int Nn) {
  int n = blockIdx.x * blockDim.x + threadIdx.x;
  if (n >= Nn) return;
  const float* row = h + (size_t)n * CLS_PAD;
  float s = 0.f, d = 0.f;
#pragma unroll
  for (int c = 0; c < CLS; ++c) {
    float v = row[c];
    s += v * a_src[c];
    d += v * a_dst[c];
  }
  als[n] = s; ald[n] = d;
}

// ---------------- edge softmax numerators + denominators ----------------
// Max-subtraction omitted: logits are O(1) here and exp(x-max) cancels exactly
// in alpha = e / sum(e), so the result is mathematically identical.

__global__ void k_edge1(const int* __restrict__ ei, int E, const float* __restrict__ als,
                        const float* __restrict__ ald, float* __restrict__ eexp,
                        float* __restrict__ denom, int EE) {
  int t = blockIdx.x * blockDim.x + threadIdx.x;
  if (t >= EE * HEADS) return;
  int e = t >> 2, hd = t & 3;
  int s, d;
  if (e < E) { s = ei[e]; d = ei[E + e]; } else { s = d = e - E; }
  float v = als[s * HEADS + hd] + ald[d * HEADS + hd];
  v = (v > 0.f) ? v : NEG_SLOPE * v;
  float ev = __expf(v);
  eexp[t] = ev;
  atomicAdd(denom + d * HEADS + hd, ev);
}

__global__ void k_edge2(const int* __restrict__ ei, int E, const float* __restrict__ als,
                        const float* __restrict__ ald, float* __restrict__ eexp,
                        float* __restrict__ denom, int EE) {
  int e = blockIdx.x * blockDim.x + threadIdx.x;
  if (e >= EE) return;
  int s, d;
  if (e < E) { s = ei[e]; d = ei[E + e]; } else { s = d = e - E; }
  float v = als[s] + ald[d];
  v = (v > 0.f) ? v : NEG_SLOPE * v;
  float ev = __expf(v);
  eexp[e] = ev;
  atomicAdd(denom + d, ev);
}

__global__ void k_invden(float* __restrict__ denom, int n) {
  int t = blockIdx.x * blockDim.x + threadIdx.x;
  if (t < n) denom[t] = 1.0f / (denom[t] + 1e-16f);
}

// ---------------- weighted scatter-sum ----------------

__global__ void k_scatter1(const int* __restrict__ ei, int E, const float* __restrict__ h,
                           const float* __restrict__ eexp, const float* __restrict__ invden,
                           float* __restrict__ out, int EE) {
  int t = blockIdx.x * blockDim.x + threadIdx.x;
  if (t >= EE * 32) return;
  int e = t >> 5, ch = t & 31;
  int c4 = ch * 4, hd = ch >> 3;           // 4 floats per thread; head = c4/32
  int s, d;
  if (e < E) { s = ei[e]; d = ei[E + e]; } else { s = d = e - E; }
  float alpha = eexp[e * HEADS + hd] * invden[d * HEADS + hd];
  const float4 hv = *(const float4*)(h + (size_t)s * F_IN + c4);
  float* o = out + (size_t)d * F_IN + c4;
  atomicAdd(o + 0, hv.x * alpha);
  atomicAdd(o + 1, hv.y * alpha);
  atomicAdd(o + 2, hv.z * alpha);
  atomicAdd(o + 3, hv.w * alpha);
}

__global__ void k_scatter2(const int* __restrict__ ei, int E, const float* __restrict__ h,
                           const float* __restrict__ eexp, const float* __restrict__ invden,
                           float* __restrict__ out, int EE) {
  int t = blockIdx.x * blockDim.x + threadIdx.x;
  if (t >= EE * 10) return;                 // CLS=40 -> 10 float4 chunks
  int e = t / 10, ch = t % 10;
  int c4 = ch * 4;
  int s, d;
  if (e < E) { s = ei[e]; d = ei[E + e]; } else { s = d = e - E; }
  float alpha = eexp[e] * invden[d];
  const float4 hv = *(const float4*)(h + (size_t)s * CLS_PAD + c4);
  float* o = out + (size_t)d * CLS + c4;
  atomicAdd(o + 0, hv.x * alpha);
  atomicAdd(o + 1, hv.y * alpha);
  atomicAdd(o + 2, hv.z * alpha);
  atomicAdd(o + 3, hv.w * alpha);
}

// ---------------- layer-1 epilogue: +b1, ReLU, cast to f16 for GEMM2 ----------------

__global__ void k_finish1(const float* __restrict__ out1, const float* __restrict__ b1,
                          _Float16* __restrict__ xh2, int n) {
  int t = blockIdx.x * blockDim.x + threadIdx.x;
  if (t >= n) return;
  float v = out1[t] + b1[t & (F_IN - 1)];
  xh2[t] = (_Float16)(v > 0.f ? v : 0.f);
}

// ---------------- final: +b2 then log_softmax over 40 classes ----------------

__global__ void k_final(const float* __restrict__ out2, const float* __restrict__ b2,
                        float* __restrict__ out, int Nn) {
  int n = blockIdx.x * blockDim.x + threadIdx.x;
  if (n >= Nn) return;
  const float* row = out2 + (size_t)n * CLS;
  float m = -3.0e38f;
#pragma unroll
  for (int c = 0; c < CLS; ++c) { float v = row[c] + b2[c]; m = v > m ? v : m; }
  float sum = 0.f;
#pragma unroll
  for (int c = 0; c < CLS; ++c) sum += __expf(row[c] + b2[c] - m);
  float lse = m + logf(sum);
  float* o = out + (size_t)n * CLS;
#pragma unroll
  for (int c = 0; c < CLS; ++c) o[c] = row[c] + b2[c] - lse;
}

// ---------------- host ----------------

extern "C" void kernel_launch(void* const* d_in, const int* in_sizes, int n_in,
                              void* d_out, int out_size, void* d_ws, size_t ws_size,
                              hipStream_t stream) {
  const float* x   = (const float*)d_in[0];
  const int*   ei  = (const int*)  d_in[1];
  const float* W1  = (const float*)d_in[2];
  const float* as1 = (const float*)d_in[3];
  const float* ad1 = (const float*)d_in[4];
  const float* b1  = (const float*)d_in[5];
  const float* W2  = (const float*)d_in[6];
  const float* as2 = (const float*)d_in[7];
  const float* ad2 = (const float*)d_in[8];
  const float* b2  = (const float*)d_in[9];
  float* out = (float*)d_out;

  const int Nn = in_sizes[0] / F_IN;
  const int E  = in_sizes[1] / 2;
  const int EE = E + Nn;

  // workspace carve-out (256B aligned)
  char* ws = (char*)d_ws;
  size_t o = 0;
  auto alloc = [&](size_t bytes) -> void* {
    void* p = ws + o;
    o = (o + bytes + 255) & ~(size_t)255;
    return p;
  };
  _Float16* xh     = (_Float16*)alloc((size_t)Nn * F_IN * 2);
  _Float16* W1T    = (_Float16*)alloc((size_t)F_IN * F_IN * 2);
  _Float16* W2T    = (_Float16*)alloc((size_t)CLS_PAD * F_IN * 2);
  float*    h1     = (float*)   alloc((size_t)Nn * F_IN * 4);
  float*    als1   = (float*)   alloc((size_t)Nn * HEADS * 4);
  float*    ald1   = (float*)   alloc((size_t)Nn * HEADS * 4);
  float*    den1   = (float*)   alloc((size_t)Nn * HEADS * 4);
  float*    eexp1  = (float*)   alloc((size_t)EE * HEADS * 4);
  float*    out1   = (float*)   alloc((size_t)Nn * F_IN * 4);
  _Float16* xh2    = (_Float16*)alloc((size_t)Nn * F_IN * 2);
  float*    h2     = (float*)   alloc((size_t)Nn * CLS_PAD * 4);
  float*    als2   = (float*)   alloc((size_t)Nn * 4);
  float*    ald2   = (float*)   alloc((size_t)Nn * 4);
  float*    den2   = (float*)   alloc((size_t)Nn * 4);
  float*    eexp2  = (float*)   alloc((size_t)EE * 4);
  float*    out2   = (float*)   alloc((size_t)Nn * CLS * 4);

  const int mtiles = (Nn + 15) / 16;

  // accumulators must be zero every call (graph replay safe: async memset on stream)
  hipMemsetAsync(den1, 0, (size_t)Nn * HEADS * 4, stream);
  hipMemsetAsync(out1, 0, (size_t)Nn * F_IN * 4, stream);
  hipMemsetAsync(den2, 0, (size_t)Nn * 4, stream);
  hipMemsetAsync(out2, 0, (size_t)Nn * CLS * 4, stream);

  auto grid = [](long long n) { return dim3((unsigned)((n + TPB - 1) / TPB)); };

  // precision prep
  k_cvt_f16<<<grid((long long)Nn * F_IN), TPB, 0, stream>>>(x, xh, Nn * F_IN);
  k_packWT<<<grid(F_IN * F_IN), TPB, 0, stream>>>(W1, W1T, F_IN, F_IN, F_IN);
  k_packWT<<<grid(F_IN * CLS_PAD), TPB, 0, stream>>>(W2, W2T, F_IN, CLS, CLS_PAD);

  // layer 1: h1 = xh @ W1  (one wave per 16-row strip, 8 n-tiles, 32 WMMAs/wave)
  k_wmma_gemm<F_IN / 16><<<grid((long long)mtiles * 32), TPB, 0, stream>>>(
      xh, W1T, h1, F_IN, Nn, mtiles);
  k_att1<<<grid((long long)Nn * HEADS), TPB, 0, stream>>>(h1, as1, ad1, als1, ald1, Nn);
  k_edge1<<<grid((long long)EE * HEADS), TPB, 0, stream>>>(ei, E, als1, ald1, eexp1, den1, EE);
  k_invden<<<grid((long long)Nn * HEADS), TPB, 0, stream>>>(den1, Nn * HEADS);
  k_scatter1<<<grid((long long)EE * 32), TPB, 0, stream>>>(ei, E, h1, eexp1, den1, out1, EE);
  k_finish1<<<grid((long long)Nn * F_IN), TPB, 0, stream>>>(out1, b1, xh2, Nn * F_IN);

  // layer 2: h2 = xh2 @ W2 (3 n-tiles, 12 WMMAs/wave)
  k_wmma_gemm<CLS_PAD / 16><<<grid((long long)mtiles * 32), TPB, 0, stream>>>(
      xh2, W2T, h2, CLS_PAD, Nn, mtiles);
  k_att2<<<grid(Nn), TPB, 0, stream>>>(h2, as2, ad2, als2, ald2, Nn);
  k_edge2<<<grid(EE), TPB, 0, stream>>>(ei, E, als2, ald2, eexp2, den2, EE);
  k_invden<<<grid(Nn), TPB, 0, stream>>>(den2, Nn);
  k_scatter2<<<grid((long long)EE * 10), TPB, 0, stream>>>(ei, E, h2, eexp2, den2, out2, EE);
  k_final<<<grid(Nn), TPB, 0, stream>>>(out2, b2, out, Nn);
}
